// disen_encoder_75161927680530
// MI455X (gfx1250) — compile-verified, matrix operand order checked by reading, not run on hardware
//
#include <hip/hip_runtime.h>
#include <hip/hip_bf16.h>
#include <math.h>

typedef __attribute__((ext_vector_type(16))) _Float16 v16h;
typedef __attribute__((ext_vector_type(8)))  float    v8f;

#define HID 64      // hidden dim
#define NCAP 8      // capsules
#define DD 8        // dims per capsule
#define SLOPE 0.01f
#define EPSN 1e-12f

__device__ __forceinline__ float lrelu(float v) { return v >= 0.0f ? v : SLOPE * v; }

// ---------------------------------------------------------------------------
// Pack W[K,64] (f32 row-major) into WMMA B-fragment order (f16):
// Wp[((kc*4+nt)*32+lane)*16 + i], lane holds column n = nt*16 + lane%16,
// K index = kc*32 + (lane>=16?8:0) + (i<8 ? i : i+8).  Zero-pad K tail.
// ---------------------------------------------------------------------------
__global__ void pack_w_kernel(const float* __restrict__ W, _Float16* __restrict__ Wp,
                              int K, int total) {
  int t = blockIdx.x * blockDim.x + threadIdx.x;
  if (t >= total) return;
  int i    =  t        & 15;
  int lane = (t >> 4)  & 31;
  int nt   = (t >> 9)  & 3;
  int kc   =  t >> 11;
  int n  = nt * 16 + (lane & 15);
  int kb = kc * 32 + ((lane & 16) ? 8 : 0);
  int k  = kb + (i < 8 ? i : i + 8);
  float v = (k < K) ? W[(size_t)k * HID + n] : 0.0f;
  Wp[t] = (_Float16)v;
}

// ---------------------------------------------------------------------------
// H[n,64] = leaky_relu(X[n,K] @ W + b) via v_wmma_f32_16x16x32_f16.
// 8 waves/block, each wave owns one 16-row M-tile and all 4 N-tiles.
// K-loop peeled: full 32-K chunks unconditional, one predicated tail chunk.
// ---------------------------------------------------------------------------
__device__ __forceinline__ v8f wmma_step(v16h a, const _Float16* wp, v8f c) {
  v16h b = *(const v16h*)wp;
  return __builtin_amdgcn_wmma_f32_16x16x32_f16(false, a, false, b, (short)0, c, false, false);
}

__global__ void wmma_gemm_bias_lrelu(const float* __restrict__ X,
                                     const _Float16* __restrict__ Wp,
                                     const float* __restrict__ bias,
                                     float* __restrict__ H,
                                     int N, int K) {
  const int lane  = threadIdx.x & 31;
  const int wave  = threadIdx.x >> 5;
  const int mtile = blockIdx.x * 8 + wave;
  const int mtiles = (N + 15) >> 4;
  if (mtile >= mtiles) return;

  int row = mtile * 16 + (lane & 15);
  if (row >= N) row = N - 1;                       // clamp load, guard store later
  const float* xrow = X + (size_t)row * K;
  const int hiSel = (lane & 16) ? 8 : 0;

  v8f c[4];
  c[0] = v8f{}; c[1] = v8f{}; c[2] = v8f{}; c[3] = v8f{};

  const int kfull = K >> 5;                        // full 32-wide K chunks
  for (int kc = 0; kc < kfull; ++kc) {
    const int kb = kc * 32 + hiSel;
    float4 a0 = *(const float4*)(xrow + kb);
    float4 a1 = *(const float4*)(xrow + kb + 4);
    float4 a2 = *(const float4*)(xrow + kb + 16);
    float4 a3 = *(const float4*)(xrow + kb + 20);
    v16h a;
    a[0]=(_Float16)a0.x; a[1]=(_Float16)a0.y; a[2]=(_Float16)a0.z; a[3]=(_Float16)a0.w;
    a[4]=(_Float16)a1.x; a[5]=(_Float16)a1.y; a[6]=(_Float16)a1.z; a[7]=(_Float16)a1.w;
    a[8]=(_Float16)a2.x; a[9]=(_Float16)a2.y; a[10]=(_Float16)a2.z; a[11]=(_Float16)a2.w;
    a[12]=(_Float16)a3.x; a[13]=(_Float16)a3.y; a[14]=(_Float16)a3.z; a[15]=(_Float16)a3.w;

    const _Float16* wp = Wp + ((size_t)(kc * 4) * 32 + lane) * 16;
    c[0] = wmma_step(a, wp,           c[0]);
    c[1] = wmma_step(a, wp + 32 * 16, c[1]);
    c[2] = wmma_step(a, wp + 64 * 16, c[2]);
    c[3] = wmma_step(a, wp + 96 * 16, c[3]);
  }

  if (K & 31) {                                    // predicated tail chunk (once)
    const int kb = kfull * 32 + hiSel;
    v16h a;
    #pragma unroll
    for (int i = 0; i < 16; ++i) {
      int k = kb + (i < 8 ? i : i + 8);
      a[i] = (k < K) ? (_Float16)xrow[k] : (_Float16)0.0f;
    }
    const _Float16* wp = Wp + ((size_t)(kfull * 4) * 32 + lane) * 16;
    c[0] = wmma_step(a, wp,           c[0]);
    c[1] = wmma_step(a, wp + 32 * 16, c[1]);
    c[2] = wmma_step(a, wp + 64 * 16, c[2]);
    c[3] = wmma_step(a, wp + 96 * 16, c[3]);
  }

  // Epilogue: D layout -> M = v + (lane>=16?8:0), N = nt*16 + lane%16
  const int colbase = lane & 15;
  const int rbase   = mtile * 16 + hiSel;
  #pragma unroll
  for (int nt = 0; nt < 4; ++nt) {
    const int   col = nt * 16 + colbase;
    const float bb  = bias[col];
    #pragma unroll
    for (int v = 0; v < 8; ++v) {
      int r = rbase + v;
      if (r < N) H[(size_t)r * HID + col] = lrelu(c[nt][v] + bb);
    }
  }
}

// ---------------------------------------------------------------------------
// capsnorm: L2-normalize each of 8 chunks of 8; one thread per (node, chunk).
// mode 1 additionally applies leaky_relu (end-of-layer fusion).
// ---------------------------------------------------------------------------
__global__ void capsnorm_kernel(const float* __restrict__ in, float* __restrict__ out,
                                int n, int mode) {
  int t = blockIdx.x * blockDim.x + threadIdx.x;
  int node = t >> 3, ch = t & 7;
  if (node >= n) return;
  const float* p = in + (size_t)node * HID + ch * DD;
  float4 v0 = *(const float4*)p;
  float4 v1 = *(const float4*)(p + 4);
  float ss = v0.x*v0.x + v0.y*v0.y + v0.z*v0.z + v0.w*v0.w
           + v1.x*v1.x + v1.y*v1.y + v1.z*v1.z + v1.w*v1.w;
  float inv = 1.0f / fmaxf(sqrtf(ss), EPSN);
  float o[8] = { v0.x*inv, v0.y*inv, v0.z*inv, v0.w*inv,
                 v1.x*inv, v1.y*inv, v1.z*inv, v1.w*inv };
  float* q = out + (size_t)node * HID + ch * DD;
  if (mode) {
    #pragma unroll
    for (int i = 0; i < 8; ++i) o[i] = lrelu(o[i]);
  }
  float4 w0 = { o[0], o[1], o[2], o[3] };
  float4 w1 = { o[4], o[5], o[6], o[7] };
  *(float4*)q       = w0;
  *(float4*)(q + 4) = w1;
}

__global__ void copy_f4_kernel(const float4* __restrict__ in, float4* __restrict__ out, int n4) {
  int t = blockIdx.x * blockDim.x + threadIdx.x;
  if (t < n4) out[t] = in[t];
}

// ---------------------------------------------------------------------------
// One routing iteration over edges. 16 lanes per edge; lane j owns 4 dims
// (half of capsule j/2). Capsule dot via shfl_xor(1); softmax over 8 capsules
// via shfl_xor(2/4/8) tree (pairs hold duplicates so {2,4,8} sums each
// capsule exactly once). Scatter-add msg into acc (pre-initialized to xn).
// Node tables (12.8 MB each) live in L2 -> gathers stay on-chip.
// ---------------------------------------------------------------------------
__global__ void edge_route_kernel(const float* __restrict__ xn, const float* __restrict__ u,
                                  const int* __restrict__ src, const int* __restrict__ trg,
                                  float* __restrict__ acc, int m) {
  int t = blockIdx.x * blockDim.x + threadIdx.x;
  int e = t >> 4, sub = t & 15;
  if (e >= m) return;
  int s = src[e], g = trg[e];

  float4 z  = *(const float4*)(xn + (size_t)s * HID + sub * 4);
  float4 uu = *(const float4*)(u  + (size_t)g * HID + sub * 4);

  float d = z.x*uu.x + z.y*uu.y + z.z*uu.z + z.w*uu.w;
  d += __shfl_xor(d, 1, 32);                 // full capsule dot, duplicated in pair

  float pt = d;                              // TAU == 1
  float mx = pt;
  mx = fmaxf(mx, __shfl_xor(mx, 2, 32));
  mx = fmaxf(mx, __shfl_xor(mx, 4, 32));
  mx = fmaxf(mx, __shfl_xor(mx, 8, 32));
  float ee = __expf(pt - mx);
  float sm = ee;
  sm += __shfl_xor(sm, 2, 32);
  sm += __shfl_xor(sm, 4, 32);
  sm += __shfl_xor(sm, 8, 32);
  float prob = ee / sm;

  float* ap = acc + (size_t)g * HID + sub * 4;
  atomicAdd(ap + 0, z.x * prob);
  atomicAdd(ap + 1, z.y * prob);
  atomicAdd(ap + 2, z.z * prob);
  atomicAdd(ap + 3, z.w * prob);
}

// ---------------------------------------------------------------------------
extern "C" void kernel_launch(void* const* d_in, const int* in_sizes, int n_in,
                              void* d_out, int out_size, void* d_ws, size_t ws_size,
                              hipStream_t stream) {
  const float* X  = (const float*)d_in[0];
  const int*   ST = (const int*)  d_in[1];
  const float* W  = (const float*)d_in[2];
  const float* B  = (const float*)d_in[3];

  const int K = in_sizes[2] / HID;           // 500
  const int n = in_sizes[0] / K;             // 50000
  const int m = in_sizes[1] / 2;             // 500000
  const int* src = ST;
  const int* trg = ST + m;

  const size_t nd = (size_t)n * HID;
  const int kchunks = (K + 31) >> 5;
  const int wpTotal = kchunks * 4 * 32 * 16;

  float*     XN  = (float*)d_ws;
  float*     U   = XN + nd;
  float*     ACC = U + nd;
  _Float16*  Wp  = (_Float16*)(ACC + nd);
  if (ws_size < 3 * nd * sizeof(float) + (size_t)wpTotal * sizeof(_Float16)) return;

  float* H = (float*)d_out;                  // activation buffer doubles as output

  // 1) pack W into B-fragment order
  pack_w_kernel<<<(wpTotal + 255) / 256, 256, 0, stream>>>(W, Wp, K, wpTotal);

  // 2) H = leaky_relu(X @ W + b) via WMMA
  {
    int mtiles = (n + 15) >> 4;
    int blocks = (mtiles + 7) >> 3;
    wmma_gemm_bias_lrelu<<<blocks, 256, 0, stream>>>(X, Wp, B, H, n, K);
  }

  const int csGrid   = (int)((nd / DD + 255) / 256);   // n*8 threads
  const int cpGrid   = (int)((nd / 4 + 255) / 256);    // float4 copy
  const int edGrid   = (int)(((size_t)m * 16 + 255) / 256);

  // 3) two routing layers
  for (int layer = 0; layer < 2; ++layer) {
    capsnorm_kernel<<<csGrid, 256, 0, stream>>>(H, XN, n, 0);
    const float* uPtr = XN;                  // u starts as capsnorm(x)
    for (int it = 0; it < 3; ++it) {
      copy_f4_kernel<<<cpGrid, 256, 0, stream>>>((const float4*)XN, (float4*)ACC, (int)(nd / 4));
      edge_route_kernel<<<edGrid, 256, 0, stream>>>(XN, uPtr, src, trg, ACC, m);
      if (it == 2) {
        capsnorm_kernel<<<csGrid, 256, 0, stream>>>(ACC, H, n, 1);  // fused leaky_relu
      } else {
        capsnorm_kernel<<<csGrid, 256, 0, stream>>>(ACC, U, n, 0);
        uPtr = U;
      }
    }
  }
}
